// TensorProductConv_7275674599931
// MI455X (gfx1250) — matrix operand order repaired and checked by reading, not compiled, not run on hardware
//
#include <hip/hip_runtime.h>
#include <math.h>

typedef __attribute__((ext_vector_type(16))) _Float16 v16h;
typedef __attribute__((ext_vector_type(8)))  _Float16 v8h;
typedef __attribute__((ext_vector_type(8)))  float    v8f;
typedef __attribute__((ext_vector_type(4)))  float    v4f;

#define N_NODES 50000
#define N_EDGES 500000
#define HIDDEN  64
#define NB      32
#define TILES   (N_EDGES / 16)   // 31250, exact

// SiLU with fast reciprocal: divisor 1+exp(-x) is in (1, inf) -> no edge
// cases; v_rcp_f32 (~1 ulp) is plenty for an f16 WMMA pipeline and avoids
// the ~8-instruction IEEE division expansion on the layer1->layer2 path.
__device__ __forceinline__ float silu_f(float x) {
    return x * __builtin_amdgcn_rcpf(1.0f + __expf(-x));
}

__device__ __forceinline__ void atomic_add_f32(float* p, float v) {
    (void)__hip_atomic_fetch_add(p, v, __ATOMIC_RELAXED, __HIP_MEMORY_SCOPE_AGENT);
}

// Compute one 16-edge tile of the MLP: silu(A @ W1 + b1) @ W2 + b2.
// a_el : A-fragment of edge_length tile (16x32 f16)
// w1   : LDS, f16, transposed layout  w1[col*32 + k], col<64, k<32
// w2   : LDS, f16, transposed layout  w2[col*64 + k], col<64, k<64
// hbuf : per-wave LDS scratch, 16*64 halfs (row-major h[m*64 + col])
// cs   : out, 4 C-fragments; cs[t][v] = weight(edge m = v+8*grp, hidden 16*t + n)
__device__ __forceinline__ void edge_mlp(v16h a_el,
                                         const _Float16* __restrict__ w1,
                                         const _Float16* __restrict__ w2,
                                         const float* __restrict__ b1,
                                         const float* __restrict__ b2,
                                         _Float16* __restrict__ hbuf,
                                         int lane, v8f cs[4])
{
    const int n   = lane & 15;
    const int grp = lane >> 4;

    // ---- layer 1: [16x32] @ [32x64] + b1, SiLU, stage to LDS ----
    asm volatile("" ::: "memory");   // order vs any prior use of hbuf (WAR)
    #pragma unroll
    for (int t = 0; t < 4; ++t) {
        // B-fragment 32x16 f16: lane group g holds K = 16*g + j (contiguous)
        v16h b = *(const v16h*)(w1 + (16 * t + n) * NB + 16 * grp);
        float bb = b1[16 * t + n];
        v8f c;
        #pragma unroll
        for (int v = 0; v < 8; ++v) c[v] = bb;
        c = __builtin_amdgcn_wmma_f32_16x16x32_f16(false, a_el, false, b,
                                                   (short)0, c, false, false);
        #pragma unroll
        for (int v = 0; v < 8; ++v) {
            float hv = silu_f(c[v]);
            hbuf[(v + 8 * grp) * HIDDEN + 16 * t + n] = (_Float16)hv;
        }
    }
    asm volatile("s_wait_dscnt 0" ::: "memory");   // cross-lane LDS visibility

    // ---- re-read h as two A-fragments (16x32 each, K-chunks 0..31 / 32..63) ----
    v16h ha0, ha1;
    {
        const _Float16* rp = hbuf + (lane & 15) * HIDDEN;
        // A 16x32 layout: grp g: halfs 0-15 -> k = 16g+0..15 (contiguous, matches B)
        v8h l0 = *(const v8h*)(rp + 16 * grp);
        v8h h0 = *(const v8h*)(rp + 8 + 16 * grp);
        v8h l1 = *(const v8h*)(rp + 32 + 16 * grp);
        v8h h1 = *(const v8h*)(rp + 40 + 16 * grp);
        #pragma unroll
        for (int i = 0; i < 8; ++i) {
            ha0[i] = l0[i]; ha0[8 + i] = h0[i];
            ha1[i] = l1[i]; ha1[8 + i] = h1[i];
        }
    }

    // ---- layer 2: [16x64] @ [64x64] + b2 (two chained K=32 WMMAs) ----
    #pragma unroll
    for (int t = 0; t < 4; ++t) {
        float bb = b2[16 * t + n];
        v8f c;
        #pragma unroll
        for (int v = 0; v < 8; ++v) c[v] = bb;
        v16h bk0 = *(const v16h*)(w2 + (16 * t + n) * HIDDEN + 16 * grp);
        v16h bk1 = *(const v16h*)(w2 + (16 * t + n) * HIDDEN + 32 + 16 * grp);
        c = __builtin_amdgcn_wmma_f32_16x16x32_f16(false, ha0, false, bk0,
                                                   (short)0, c, false, false);
        c = __builtin_amdgcn_wmma_f32_16x16x32_f16(false, ha1, false, bk1,
                                                   (short)0, c, false, false);
        cs[t] = c;
    }
}

__launch_bounds__(256)
__global__ void tpconv_fused_kernel(const float* __restrict__ scalar,
                                    const float* __restrict__ vector,
                                    const float* __restrict__ edge_length,
                                    const int*   __restrict__ edge_index,
                                    const float* __restrict__ sw1, const float* __restrict__ sb1,
                                    const float* __restrict__ sw2, const float* __restrict__ sb2,
                                    const float* __restrict__ vw1, const float* __restrict__ vb1,
                                    const float* __restrict__ vw2, const float* __restrict__ vb2,
                                    float* __restrict__ out_scalar,
                                    float* __restrict__ out_vector)
{
    __shared__ _Float16 w1t[2][HIDDEN * NB];      // [mlp][col*32 + k]  8 KB
    __shared__ _Float16 w2t[2][HIDDEN * HIDDEN];  // [mlp][col*64 + k] 16 KB
    __shared__ float    bias[4][HIDDEN];          // sb1, sb2, vb1, vb2 1 KB
    __shared__ _Float16 hstage[8][16 * HIDDEN];   // per-wave h scratch 16 KB

    const int tid = threadIdx.x;

    // ---- stage weights (transposed, f16) + biases into LDS ----
    for (int i = tid; i < NB * HIDDEN; i += 256) {
        int k = i >> 6, c = i & 63;           // sw1 is [k][c] row-major, coalesced read
        w1t[0][c * NB + k] = (_Float16)sw1[i];
        w1t[1][c * NB + k] = (_Float16)vw1[i];
    }
    for (int i = tid; i < HIDDEN * HIDDEN; i += 256) {
        int k = i >> 6, c = i & 63;
        w2t[0][c * HIDDEN + k] = (_Float16)sw2[i];
        w2t[1][c * HIDDEN + k] = (_Float16)vw2[i];
    }
    if (tid < HIDDEN) {
        bias[0][tid] = sb1[tid];
        bias[1][tid] = sb2[tid];
        bias[2][tid] = vb1[tid];
        bias[3][tid] = vb2[tid];
    }
    __syncthreads();

    const int wave = tid >> 5;
    const int lane = tid & 31;
    const int tile = blockIdx.x * 8 + wave;
    if (tile >= TILES) return;                // wave-uniform; EXEC stays all-ones

    const int e0  = tile * 16;
    const int m   = lane & 15;
    const int grp = lane >> 4;
    const int n   = m;

    // ---- A-fragment of edge_length tile (f32 -> f16) ----
    v16h a_el;
    {
        const float* rowp = edge_length + (size_t)(e0 + m) * NB;   // 128B-aligned row
        v4f x0 = *(const v4f*)(rowp + 16 * grp);
        v4f x1 = *(const v4f*)(rowp + 16 * grp + 4);
        v4f y0 = *(const v4f*)(rowp + 16 * grp + 8);
        v4f y1 = *(const v4f*)(rowp + 16 * grp + 12);
        #pragma unroll
        for (int i = 0; i < 4; ++i) {
            a_el[i]      = (_Float16)x0[i];
            a_el[4 + i]  = (_Float16)x1[i];
            a_el[8 + i]  = (_Float16)y0[i];
            a_el[12 + i] = (_Float16)y1[i];
        }
    }

    v8f cs[4];

    // ================= scalar branch =================
    edge_mlp(a_el, w1t[0], w2t[0], bias[0], bias[1], hstage[wave], lane, cs);

    #pragma unroll
    for (int v = 0; v < 8; ++v) {
        const int e = e0 + v + 8 * grp;
        const int r = edge_index[e];              // 2 distinct addrs/wave: broadcast
        const int c = edge_index[N_EDGES + e];
        const float* src = scalar + (size_t)c * HIDDEN;
        float* dst       = out_scalar + (size_t)r * HIDDEN;
        #pragma unroll
        for (int t = 0; t < 4; ++t) {
            const int h = 16 * t + n;
            atomic_add_f32(&dst[h], src[h] * cs[t][v]);
        }
    }

    // ================= vector branch =================
    edge_mlp(a_el, w1t[1], w2t[1], bias[2], bias[3], hstage[wave], lane, cs);

    #pragma unroll
    for (int v = 0; v < 8; ++v) {
        const int e = e0 + v + 8 * grp;
        const int r = edge_index[e];
        const int c = edge_index[N_EDGES + e];
        const float* src = vector + (size_t)c * 3 * HIDDEN;
        float* dst       = out_vector + (size_t)r * 3 * HIDDEN;
        #pragma unroll
        for (int t = 0; t < 4; ++t) {
            const int h = 16 * t + n;
            const float w = cs[t][v];
            #pragma unroll
            for (int d = 0; d < 3; ++d) {
                atomic_add_f32(&dst[d * HIDDEN + h], src[d * HIDDEN + h] * w);
            }
        }
    }
}

extern "C" void kernel_launch(void* const* d_in, const int* in_sizes, int n_in,
                              void* d_out, int out_size, void* d_ws, size_t ws_size,
                              hipStream_t stream) {
    const float* scalar      = (const float*)d_in[0];
    const float* vector      = (const float*)d_in[1];
    // d_in[2] = edge_sh (unused by reference)
    const float* edge_length = (const float*)d_in[3];
    const int*   edge_index  = (const int*)  d_in[4];
    const float* sw1 = (const float*)d_in[5];
    const float* sb1 = (const float*)d_in[6];
    const float* sw2 = (const float*)d_in[7];
    const float* sb2 = (const float*)d_in[8];
    const float* vw1 = (const float*)d_in[9];
    const float* vb1 = (const float*)d_in[10];
    const float* vw2 = (const float*)d_in[11];
    const float* vb2 = (const float*)d_in[12];

    float* out_scalar = (float*)d_out;
    float* out_vector = out_scalar + (size_t)N_NODES * HIDDEN;

    // Outputs are accumulated with atomics -> zero first (graph-capture safe).
    hipMemsetAsync(d_out, 0, (size_t)out_size * sizeof(float), stream);

    const int blocks = (TILES + 7) / 8;   // 8 waves (tiles) per 256-thread block
    tpconv_fused_kernel<<<blocks, 256, 0, stream>>>(
        scalar, vector, edge_length, edge_index,
        sw1, sb1, sw2, sb2, vw1, vb1, vw2, vb2,
        out_scalar, out_vector);
}
